// EC_Encoder_62337155334742
// MI455X (gfx1250) — matrix operand order repaired and checked by reading, not compiled, not run on hardware
//
#include <hip/hip_runtime.h>
#include <hip/hip_bf16.h>
#include <cmath>

typedef __attribute__((ext_vector_type(16))) _Float16 v16h;
typedef __attribute__((ext_vector_type(8)))  float    v8f;
typedef __attribute__((ext_vector_type(4)))  unsigned int u32x4;
typedef __attribute__((ext_vector_type(8)))  int      i32x8;
typedef __attribute__((ext_vector_type(4)))  int      i32x4;

#define ACT_NONE 0
#define ACT_GELU 1

#if defined(__HIP_DEVICE_COMPILE__) && defined(__has_builtin)
#if __has_builtin(__builtin_amdgcn_tensor_load_to_lds)
#define USE_TDM 1
#endif
#endif
#ifndef USE_TDM
#define USE_TDM 0
#endif

// ---------------------------------------------------------------------------
// TDM: 2D f16 tile load (tile_x cols x tile_y rows) into LDS.
// D# layout per CDNA5 ISA (group0 128b, group1 256b).
// ---------------------------------------------------------------------------
#if USE_TDM
__device__ __forceinline__ void tdm_load_tile_f16(
    unsigned ldsOff, const void* gaddr,
    unsigned tile_x, unsigned tile_y,
    unsigned long tensor_x, unsigned long tensor_y, unsigned long stride_x)
{
    unsigned long ga = (unsigned long)gaddr;
    u32x4 g0;
    g0[0] = 1u;                                            // count=1, user mode
    g0[1] = ldsOff;                                        // lds_addr [63:32]
    g0[2] = (unsigned)(ga & 0xFFFFFFFFu);                  // global_addr[31:0]
    g0[3] = (unsigned)((ga >> 32) & 0x01FFFFFFu) | (2u << 30); // addr[56:32] | type=2
    i32x8 g1;
    g1[0] = (int)(1u << 16);                               // data_size=1 -> 2 bytes
    g1[1] = (int)((tensor_x & 0xFFFFu) << 16);             // tensor_dim0[15:0]
    g1[2] = (int)(((tensor_x >> 16) & 0xFFFFu) | ((tensor_y & 0xFFFFu) << 16));
    g1[3] = (int)(((tensor_y >> 16) & 0xFFFFu) | ((tile_x & 0xFFFFu) << 16)); // tile_dim0
    g1[4] = (int)(tile_y & 0xFFFFu);                       // tile_dim1 (tile_dim2=0)
    g1[5] = (int)(stride_x & 0xFFFFFFFFu);                 // tensor_dim0_stride[31:0]
    g1[6] = (int)((stride_x >> 32) & 0xFFFFu);             // stride[47:32]
    g1[7] = 0;
    i32x4 z4 = {0, 0, 0, 0};
#if __clang_major__ >= 23
    i32x8 z8 = {0, 0, 0, 0, 0, 0, 0, 0};
    __builtin_amdgcn_tensor_load_to_lds(g0, g1, z4, z4, z8, 0);
#else
    __builtin_amdgcn_tensor_load_to_lds(g0, g1, z4, z4, 0);
#endif
}
#endif

// ---------------------------------------------------------------------------
// WMMA GEMM: C[M,N] = A[M,K](f16 row-major) * Bp(packed f16) + bias (+res,+GELU)
// One 16x16 tile per wave, 8 waves/block. K unrolled x2 (two v_wmma per iter).
// TDM path: per-wave double-buffered 16x64 A tile staged through LDS; next
// tile's tensor_load_to_lds is issued before s_wait_tensorcnt<=1 on the
// current one (TDM completions are in-order per wave).
// Packed B: for (tileN t, ktile kt, lane, e): Bp[((t*(K/32)+kt)*32+lane)*16+e]
//   = B[kt*32 + ((lane&16)?8:0) + (e&7) + ((e&8)?16:0)][t*16 + (lane&15)]  -> lane reads 32B blocks.
// A fragment per ISA 7.12.2: lanes 0-15 Kbase 0, lanes 16-31 Kbase 8;
//   halves 0..7 -> K+0..7, halves 8..15 -> K+16..23.
// ---------------------------------------------------------------------------
__global__ __launch_bounds__(256)
void gemm_wmma(const _Float16* __restrict__ A, const _Float16* __restrict__ Bp,
               const float* __restrict__ bias, const float* __restrict__ resid,
               float* __restrict__ C, int M, int N, int K, int act)
{
    const int lane = threadIdx.x & 31;
    const int wave = threadIdx.x >> 5;
    const int row0 = (blockIdx.y * 8 + wave) * 16;
    const int col0 = blockIdx.x * 16;
    if (row0 >= M || col0 >= N) return;   // uniform per wave: EXEC all-1 inside

    const int nB = col0 + (lane & 15);
    const int kOffA = (lane & 16) ? 8 : 0;
    const _Float16* pbBase = Bp + ((long)blockIdx.x * (K >> 5)) * 512 + lane * 16;

    v8f acc = {};
#if USE_TDM
    // Double-buffered per-wave 16x64 A tiles staged via the Tensor Data Mover.
    __shared__ _Float16 ldsA[8][2][16 * 64];
    const unsigned lds0 = (unsigned)(unsigned long)(uintptr_t)(&ldsA[wave][0][0]);
    const unsigned lds1 = (unsigned)(unsigned long)(uintptr_t)(&ldsA[wave][1][0]);
    const _Float16* gA = A + (long)row0 * K;
    tdm_load_tile_f16(lds0, gA, 64u, 16u, (unsigned long)K, 16ul, (unsigned long)K);
    for (int k0 = 0; k0 < K; k0 += 64) {
        const int cur = (k0 >> 6) & 1;
        if (k0 + 64 < K) {
            tdm_load_tile_f16(cur ? lds0 : lds1, gA + k0 + 64, 64u, 16u,
                              (unsigned long)K, 16ul, (unsigned long)K);
            __builtin_amdgcn_s_wait_tensorcnt((short)1);   // older transfer done
        } else {
            __builtin_amdgcn_s_wait_tensorcnt((short)0);
        }
        // TDM wrote LDS behind the compiler's back: force re-read of the tile.
        asm volatile("" ::: "memory");
        const _Float16* tile = &ldsA[wave][cur][0];
#pragma unroll
        for (int s = 0; s < 2; ++s) {
            const _Float16* pa = tile + (lane & 15) * 64 + s * 32 + kOffA;
            v16h a;
#pragma unroll
            for (int e = 0; e < 8; ++e) { a[e] = pa[e]; a[8 + e] = pa[16 + e]; }
            v16h b = *(const v16h*)(pbBase + (long)((k0 >> 5) + s) * 512);
            acc = __builtin_amdgcn_wmma_f32_16x16x32_f16(
                      false, a, false, b, (short)0, acc, false, false);
        }
    }
#else
    const int mA = row0 + (lane & 15);
    for (int k0 = 0; k0 < K; k0 += 64) {
#pragma unroll
        for (int s = 0; s < 2; ++s) {
            const _Float16* pa = A + (long)mA * K + k0 + s * 32 + kOffA;
            v16h a;
#pragma unroll
            for (int e = 0; e < 8; ++e) { a[e] = pa[e]; a[8 + e] = pa[16 + e]; }
            v16h b = *(const v16h*)(pbBase + (long)((k0 >> 5) + s) * 512);
            if (k0 + 64 < K) __builtin_prefetch(pa + 64, 0, 1);   // global_prefetch_b8
            acc = __builtin_amdgcn_wmma_f32_16x16x32_f16(
                      false, a, false, b, (short)0, acc, false, false);
        }
    }
#endif

    const int mBase = row0 + ((lane & 16) ? 8 : 0);
#pragma unroll
    for (int r = 0; r < 8; ++r) {
        int m = mBase + r;
        float v = acc[r];
        if (bias)  v += bias[nB];
        if (resid) v += resid[(long)m * N + nB];
        if (act == ACT_GELU) v = 0.5f * v * (1.0f + erff(v * 0.70710678f));
        C[(long)m * N + nB] = v;
    }
}

// ---------------------------------------------------------------------------
__global__ void cvt_f32_to_f16(const float* __restrict__ s, _Float16* __restrict__ d, long n)
{
    long i = (long)blockIdx.x * 256 + threadIdx.x;
    if (i < n) d[i] = (_Float16)s[i];
}

// f32 (KxN row-major) -> fragment-packed f16 (see gemm_wmma header comment)
__global__ void pack_b_f16(const float* __restrict__ B, _Float16* __restrict__ Bp,
                           int K, int N)
{
    long total = (long)K * N;
    long i = (long)blockIdx.x * 256 + threadIdx.x;
    if (i >= total) return;
    int e    = (int)(i & 15);
    int lane = (int)((i >> 4) & 31);
    long pkt = i >> 9;
    int KT = K >> 5;
    int kt = (int)(pkt % KT);
    int t  = (int)(pkt / KT);
    int kk = kt * 32 + ((lane & 16) ? 8 : 0) + (e & 7) + ((e & 8) ? 16 : 0);
    int n  = t * 16 + (lane & 15);
    Bp[i] = (_Float16)B[(long)kk * N + n];
}

__global__ void fill_const(float* p, float v, long n)
{
    long i = (long)blockIdx.x * 256 + threadIdx.x;
    if (i < n) p[i] = v;
}

// ---------------------------------------------------------------------------
// token conv-embed (circular, k=3, Ein=7) + positional + 4 temporal sinusoids
__device__ __forceinline__ float sinus_val(int pos, int d, int dm)
{
    float div = __expf((float)(2 * (d / 2)) * (-9.210340372f / (float)dm)); // ln(1e4)
    float ang = (float)pos * div;
    return (d & 1) ? __cosf(ang) : __sinf(ang);
}

__global__ void embed_kernel(const float* __restrict__ x_enc, const int* __restrict__ x_mark,
                             const float* __restrict__ token_w, float* __restrict__ x,
                             int B, int L, int Ein, int dm)
{
    long i = (long)blockIdx.x * 256 + threadIdx.x;
    long total = (long)B * L * dm;
    if (i >= total) return;
    int d = (int)(i % dm);
    int l = (int)((i / dm) % L);
    int b = (int)(i / ((long)dm * L));
    float acc = 0.f;
    for (int t = 0; t < 3; ++t) {
        int ls = (l - 1 + t + L) % L;
        const float* px = x_enc + ((long)b * L + ls) * Ein;
        const float* pw = token_w + ((long)t * Ein) * dm + d;
        for (int c = 0; c < Ein; ++c) acc += px[c] * pw[(long)c * dm];
    }
    acc += sinus_val(l, d, dm);
    const int* pm = x_mark + ((long)b * L + l) * 4;
    acc += sinus_val(pm[3], d, dm) + sinus_val(pm[2], d, dm)
         + sinus_val(pm[1], d, dm) + sinus_val(pm[0], d, dm);
    x[i] = acc;
}

// ---------------------------------------------------------------------------
// (residual-add +) LayerNorm, one 256-thread block per row of dm elements
__global__ __launch_bounds__(256)
void add_layernorm(const float* __restrict__ x, const float* __restrict__ r,
                   const float* __restrict__ g, const float* __restrict__ be,
                   float* __restrict__ out, int dm)
{
    __shared__ float red[256];
    long row = blockIdx.x;
    const float* px = x + row * dm;
    const float* pr = r ? r + row * dm : nullptr;
    float s = 0.f;
    for (int d = threadIdx.x; d < dm; d += 256) s += px[d] + (pr ? pr[d] : 0.f);
    red[threadIdx.x] = s; __syncthreads();
    for (int o = 128; o > 0; o >>= 1) { if (threadIdx.x < o) red[threadIdx.x] += red[threadIdx.x + o]; __syncthreads(); }
    float mean = red[0] / dm; __syncthreads();
    float s2 = 0.f;
    for (int d = threadIdx.x; d < dm; d += 256) {
        float v = px[d] + (pr ? pr[d] : 0.f) - mean; s2 += v * v;
    }
    red[threadIdx.x] = s2; __syncthreads();
    for (int o = 128; o > 0; o >>= 1) { if (threadIdx.x < o) red[threadIdx.x] += red[threadIdx.x + o]; __syncthreads(); }
    float rstd = rsqrtf(red[0] / dm + 1e-5f);
    for (int d = threadIdx.x; d < dm; d += 256) {
        float v = px[d] + (pr ? pr[d] : 0.f);
        out[row * dm + d] = (v - mean) * rstd * g[d] + be[d];
    }
}

// ---------------------------------------------------------------------------
// ProbSparse attention helpers. q/k/v live in (b,l,h,d) layout (GEMM output).
__global__ void gen_sidx(int* sidx, int L, int U, int Lk, unsigned seed)
{
    int i = blockIdx.x * 256 + threadIdx.x;
    if (i >= L * U) return;
    unsigned h = (unsigned)i * 2654435761u ^ seed;
    h ^= h >> 16; h *= 2246822519u; h ^= h >> 13; h *= 3266489917u; h ^= h >> 16;
    sidx[i] = (int)(h % (unsigned)Lk);
}

__global__ void sample_scores(const float* __restrict__ q, const float* __restrict__ k,
                              const int* __restrict__ sidx, float* __restrict__ Msc,
                              int B, int H, int L, int dh, int U)
{
    long i = (long)blockIdx.x * 256 + threadIdx.x;
    if (i >= (long)B * H * L) return;
    int l = (int)(i % L); int h = (int)((i / L) % H); int b = (int)(i / ((long)L * H));
    const float* pq = q + (((long)b * L + l) * H + h) * dh;
    float mx = -1e30f, sm = 0.f;
    for (int j = 0; j < U; ++j) {
        int ks = sidx[l * U + j];
        const float* pk = k + (((long)b * L + ks) * H + h) * dh;
        float dot = 0.f;
        for (int d = 0; d < dh; ++d) dot += pq[d] * pk[d];
        mx = fmaxf(mx, dot); sm += dot;
    }
    Msc[i] = mx - sm / (float)U;
}

__global__ __launch_bounds__(256)
void topk_kernel(const float* __restrict__ Msc, int* __restrict__ top, int L, int u)
{
    __shared__ float s[512];
    __shared__ float rv[256];
    __shared__ int   ri[256];
    int bh = blockIdx.x;
    for (int l = threadIdx.x; l < L; l += 256) s[l] = Msc[(long)bh * L + l];
    __syncthreads();
    for (int it = 0; it < u; ++it) {
        float bv = -1e30f; int bi = 0;
        for (int l = threadIdx.x; l < L; l += 256)
            if (s[l] > bv) { bv = s[l]; bi = l; }
        rv[threadIdx.x] = bv; ri[threadIdx.x] = bi; __syncthreads();
        for (int o = 128; o > 0; o >>= 1) {
            if (threadIdx.x < o) {
                if (rv[threadIdx.x + o] > rv[threadIdx.x] ||
                    (rv[threadIdx.x + o] == rv[threadIdx.x] && ri[threadIdx.x + o] < ri[threadIdx.x])) {
                    rv[threadIdx.x] = rv[threadIdx.x + o];
                    ri[threadIdx.x] = ri[threadIdx.x + o];
                }
            }
            __syncthreads();
        }
        if (threadIdx.x == 0) { top[(long)bh * u + it] = ri[0]; s[ri[0]] = -1e30f; }
        __syncthreads();
    }
}

__global__ void vmean_kernel(const float* __restrict__ v, float* __restrict__ vm,
                             int B, int H, int L, int dh)
{
    int i = blockIdx.x * 256 + threadIdx.x;
    if (i >= B * H * dh) return;
    int d = i % dh; int h = (i / dh) % H; int b = i / (dh * H);
    float s = 0.f;
    for (int l = 0; l < L; ++l) s += v[(((long)b * L + l) * H + h) * dh + d];
    vm[i] = s / (float)L;
}

__global__ void fill_ctx(float* __restrict__ ctx, const float* __restrict__ vm,
                         int B, int L, int H, int dh)
{
    long total = (long)B * L * H * dh;
    long i = (long)blockIdx.x * 256 + threadIdx.x;
    if (i >= total) return;
    int d = (int)(i % dh); int h = (int)((i / dh) % H);
    int b = (int)(i / ((long)dh * H * L));
    ctx[i] = vm[((long)b * H + h) * dh + d];
}

// one block per selected query row: scores, softmax, attn-row write, ctx scatter
__global__ __launch_bounds__(256)
void attn_rows(const float* __restrict__ q, const float* __restrict__ k,
               const float* __restrict__ v, const int* __restrict__ top,
               float* __restrict__ attn, float* __restrict__ ctx,
               int B, int H, int L, int dh, int u)
{
    __shared__ float sc[512];
    __shared__ float red[256];
    int i = blockIdx.x;
    int ui = i % u; int h = (i / u) % H; int b = i / (u * H);
    int lq = top[((long)b * H + h) * u + ui];
    const float* pq = q + (((long)b * L + lq) * H + h) * dh;
    float scale = rsqrtf((float)dh);

    float mx = -1e30f;
    for (int kk = threadIdx.x; kk < L; kk += 256) {
        const float* pk = k + (((long)b * L + kk) * H + h) * dh;
        float dot = 0.f;
        for (int d = 0; d < dh; ++d) dot += pq[d] * pk[d];
        dot *= scale; sc[kk] = dot; mx = fmaxf(mx, dot);
    }
    red[threadIdx.x] = mx; __syncthreads();
    for (int o = 128; o > 0; o >>= 1) { if (threadIdx.x < o) red[threadIdx.x] = fmaxf(red[threadIdx.x], red[threadIdx.x + o]); __syncthreads(); }
    mx = red[0]; __syncthreads();

    float s = 0.f;
    for (int kk = threadIdx.x; kk < L; kk += 256) { float e = __expf(sc[kk] - mx); sc[kk] = e; s += e; }
    red[threadIdx.x] = s; __syncthreads();
    for (int o = 128; o > 0; o >>= 1) { if (threadIdx.x < o) red[threadIdx.x] += red[threadIdx.x + o]; __syncthreads(); }
    float inv = 1.f / red[0]; __syncthreads();

    float* arow = attn + (((long)b * H + h) * L + lq) * L;
    for (int kk = threadIdx.x; kk < L; kk += 256) { float a = sc[kk] * inv; sc[kk] = a; arow[kk] = a; }
    __syncthreads();

    if (threadIdx.x < dh) {
        int d = threadIdx.x;
        float accv = 0.f;
        for (int kk = 0; kk < L; ++kk)
            accv += sc[kk] * v[(((long)b * L + kk) * H + h) * dh + d];
        ctx[(((long)b * L + lq) * H + h) * dh + d] = accv;
    }
}

// ---------------------------------------------------------------------------
// distilling: im2col (circular k=3) -> GEMM -> BN stats -> BN+ELU -> maxpool/2
__global__ void im2col3_f16(const float* __restrict__ x, _Float16* __restrict__ a,
                            int B, int L, int dm)
{
    long total = (long)B * L * 3 * dm;
    long i = (long)blockIdx.x * 256 + threadIdx.x;
    if (i >= total) return;
    int c = (int)(i % dm);
    int t = (int)((i / dm) % 3);
    long bl = i / (3l * dm);
    int l = (int)(bl % L); int b = (int)(bl / L);
    int ls = (l - 1 + t + L) % L;
    a[i] = (_Float16)x[((long)b * L + ls) * dm + c];
}

__global__ __launch_bounds__(256)
void bn_stats(const float* __restrict__ y, float* __restrict__ mu, float* __restrict__ var,
              int Mrows, int dm)
{
    __shared__ float s1[256], s2[256];
    int n = blockIdx.x;
    float a = 0.f, b = 0.f;
    for (int r = threadIdx.x; r < Mrows; r += 256) {
        float v = y[(long)r * dm + n]; a += v; b += v * v;
    }
    s1[threadIdx.x] = a; s2[threadIdx.x] = b; __syncthreads();
    for (int o = 128; o > 0; o >>= 1) {
        if (threadIdx.x < o) { s1[threadIdx.x] += s1[threadIdx.x + o]; s2[threadIdx.x] += s2[threadIdx.x + o]; }
        __syncthreads();
    }
    if (threadIdx.x == 0) {
        float m = s1[0] / Mrows;
        mu[n] = m; var[n] = s2[0] / Mrows - m * m;
    }
}

__global__ void bn_elu(float* __restrict__ y, const float* __restrict__ mu,
                       const float* __restrict__ var, const float* __restrict__ g,
                       const float* __restrict__ be, long total, int dm)
{
    long i = (long)blockIdx.x * 256 + threadIdx.x;
    if (i >= total) return;
    int n = (int)(i % dm);
    float v = (y[i] - mu[n]) * rsqrtf(var[n] + 1e-5f) * g[n] + be[n];
    y[i] = v > 0.f ? v : (__expf(v) - 1.f);
}

__global__ void maxpool_kernel(const float* __restrict__ y, float* __restrict__ x,
                               int B, int L, int dm)
{
    int Lout = L / 2;
    long total = (long)B * Lout * dm;
    long i = (long)blockIdx.x * 256 + threadIdx.x;
    if (i >= total) return;
    int d = (int)(i % dm);
    int lo = (int)((i / dm) % Lout);
    int b = (int)(i / ((long)dm * Lout));
    float m = -1e30f;
    for (int t = 0; t < 3; ++t) {
        int lp = 2 * lo - 1 + t;
        if (lp >= 0 && lp < L) m = fmaxf(m, y[((long)b * L + lp) * dm + d]);
    }
    x[i] = m;
}

__global__ void softmax_rows(const float* __restrict__ logits, float* __restrict__ out, int n)
{
    __shared__ float red[64];
    int r = blockIdx.x, t = threadIdx.x;
    float v = (t < n) ? logits[r * n + t] : -1e30f;
    red[t] = v; __syncthreads();
    for (int o = 32; o > 0; o >>= 1) { if (t < o) red[t] = fmaxf(red[t], red[t + o]); __syncthreads(); }
    float mx = red[0]; __syncthreads();
    float e = (t < n) ? __expf(v - mx) : 0.f;
    red[t] = e; __syncthreads();
    for (int o = 32; o > 0; o >>= 1) { if (t < o) red[t] += red[t + o]; __syncthreads(); }
    if (t < n) out[r * n + t] = e / red[0];
}

// ---------------------------------------------------------------------------
extern "C" void kernel_launch(void* const* d_in, const int* in_sizes, int n_in,
                              void* d_out, int out_size, void* d_ws, size_t ws_size,
                              hipStream_t stream)
{
    (void)in_sizes; (void)n_in; (void)out_size; (void)ws_size;
    const float* x_enc   = (const float*)d_in[0];
    const int*   x_mark  = (const int*)d_in[1];
    const float* token_w = (const float*)d_in[4];
    const float* Wq = (const float*)d_in[5];  const float* Bq = (const float*)d_in[6];
    const float* Wk = (const float*)d_in[7];  const float* Bk = (const float*)d_in[8];
    const float* Wv = (const float*)d_in[9];  const float* Bv = (const float*)d_in[10];
    const float* Wo = (const float*)d_in[11]; const float* Bo = (const float*)d_in[12];
    const float* ln1_g = (const float*)d_in[13]; const float* ln1_b = (const float*)d_in[14];
    const float* ln2_g = (const float*)d_in[15]; const float* ln2_b = (const float*)d_in[16];
    const float* ff1_w = (const float*)d_in[17]; const float* ff1_b = (const float*)d_in[18];
    const float* ff2_w = (const float*)d_in[19]; const float* ff2_b = (const float*)d_in[20];
    const float* conv_w = (const float*)d_in[21]; const float* conv_b = (const float*)d_in[22];
    const float* bn_g = (const float*)d_in[23]; const float* bn_b = (const float*)d_in[24];
    const float* lnf_g = (const float*)d_in[25]; const float* lnf_b = (const float*)d_in[26];
    const float* proj_w = (const float*)d_in[27]; const float* proj_b = (const float*)d_in[28];
    float* out = (float*)d_out;

    const int Bsz = 16, DM = 512, Hh = 8, DH = 64, EIN = 7, DFF = 512;
    const long MAXM = (long)Bsz * 512;

    char* ws = (char*)d_ws;
    auto alloc = [&](size_t bytes) -> char* {
        char* p = ws; ws += (bytes + 255) & ~(size_t)255; return p;
    };
    float* xbuf  = (float*)alloc(MAXM * DM * 4);
    float* rbuf  = (float*)alloc(MAXM * DM * 4);
    float* qb    = (float*)alloc(MAXM * DM * 4);
    float* kb    = (float*)alloc(MAXM * DM * 4);
    float* vb    = (float*)alloc(MAXM * DM * 4);
    float* ctx   = (float*)alloc(MAXM * DM * 4);
    _Float16* a16 = (_Float16*)alloc((size_t)MAXM * 3 * DM * 2);
    _Float16* b16 = (_Float16*)alloc((size_t)65536 * 48 * 2);  // >= 3*DM*DM*2
    int*   sidx  = (int*)alloc(512 * 35 * 4);
    float* Msc   = (float*)alloc((long)Bsz * Hh * 512 * 4);
    int*   top   = (int*)alloc((long)Bsz * Hh * 35 * 4);
    float* vmean = (float*)alloc((long)Bsz * Hh * DH * 4);
    float* mu    = (float*)alloc(DM * 4);
    float* var   = (float*)alloc(DM * 4);
    float* logits = (float*)alloc(16 * 48 * 4);

    auto cvt = [&](const float* s, _Float16* d, long n) {
        cvt_f32_to_f16<<<dim3((unsigned)((n + 255) / 256)), 256, 0, stream>>>(s, d, n);
    };
    auto packB = [&](const float* s, _Float16* d, int K, int N) {
        long total = (long)K * N;
        pack_b_f16<<<dim3((unsigned)((total + 255) / 256)), 256, 0, stream>>>(s, d, K, N);
    };
    auto gemm = [&](const _Float16* A, const _Float16* Bm, const float* bias,
                    const float* res, float* C, int M, int N, int K, int act) {
        dim3 g((unsigned)(N / 16), (unsigned)((M + 127) / 128));
        gemm_wmma<<<g, 256, 0, stream>>>(A, Bm, bias, res, C, M, N, K, act);
    };

    // embedding
    {
        long total = (long)Bsz * 512 * DM;
        embed_kernel<<<dim3((unsigned)((total + 255) / 256)), 256, 0, stream>>>(
            x_enc, x_mark, token_w, xbuf, Bsz, 512, EIN, DM);
    }

    int Lc = 512;
    float* attn_out = out + 768;  // probs(16x48) first, then attn tensors
    for (int l = 0; l < 3; ++l) {
        int Mr = Bsz * Lc;
        int U = (int)(5.0 * ceil(log((double)Lc)));
        if (U > Lc) U = Lc;

        // QKV projections (WMMA)
        cvt(xbuf, a16, (long)Mr * DM);
        packB(Wq + (long)l * DM * DM, b16, DM, DM);
        gemm(a16, b16, Bq + l * DM, nullptr, qb, Mr, DM, DM, ACT_NONE);
        packB(Wk + (long)l * DM * DM, b16, DM, DM);
        gemm(a16, b16, Bk + l * DM, nullptr, kb, Mr, DM, DM, ACT_NONE);
        packB(Wv + (long)l * DM * DM, b16, DM, DM);
        gemm(a16, b16, Bv + l * DM, nullptr, vb, Mr, DM, DM, ACT_NONE);

        // ProbSparse attention
        gen_sidx<<<dim3((unsigned)((Lc * U + 255) / 256)), 256, 0, stream>>>(
            sidx, Lc, U, Lc, 0x9E3779B9u * (unsigned)(l + 1));
        sample_scores<<<dim3((unsigned)(((long)Bsz * Hh * Lc + 255) / 256)), 256, 0, stream>>>(
            qb, kb, sidx, Msc, Bsz, Hh, Lc, DH, U);
        topk_kernel<<<dim3((unsigned)(Bsz * Hh)), 256, 0, stream>>>(Msc, top, Lc, U);
        vmean_kernel<<<dim3((unsigned)((Bsz * Hh * DH + 255) / 256)), 256, 0, stream>>>(
            vb, vmean, Bsz, Hh, Lc, DH);
        fill_ctx<<<dim3((unsigned)(((long)Mr * DM + 255) / 256)), 256, 0, stream>>>(
            ctx, vmean, Bsz, Lc, Hh, DH);
        fill_const<<<dim3((unsigned)(((long)Bsz * Hh * Lc * Lc + 255) / 256)), 256, 0, stream>>>(
            attn_out, 1.0f / (float)Lc, (long)Bsz * Hh * Lc * Lc);
        attn_rows<<<dim3((unsigned)(Bsz * Hh * U)), 256, 0, stream>>>(
            qb, kb, vb, top, attn_out, ctx, Bsz, Hh, Lc, DH, U);

        // output projection + residual LN
        cvt(ctx, a16, (long)Mr * DM);
        packB(Wo + (long)l * DM * DM, b16, DM, DM);
        gemm(a16, b16, Bo + l * DM, nullptr, rbuf, Mr, DM, DM, ACT_NONE);
        add_layernorm<<<dim3((unsigned)Mr), 256, 0, stream>>>(
            xbuf, rbuf, ln1_g + l * DM, ln1_b + l * DM, xbuf, DM);

        // FFN
        cvt(xbuf, a16, (long)Mr * DM);
        packB(ff1_w + (long)l * DM * DFF, b16, DM, DFF);
        gemm(a16, b16, ff1_b + l * DFF, nullptr, rbuf, Mr, DFF, DM, ACT_GELU);
        cvt(rbuf, a16, (long)Mr * DFF);
        packB(ff2_w + (long)l * DFF * DM, b16, DFF, DM);
        gemm(a16, b16, ff2_b + l * DM, nullptr, rbuf, Mr, DM, DFF, ACT_NONE);
        add_layernorm<<<dim3((unsigned)Mr), 256, 0, stream>>>(
            xbuf, rbuf, ln2_g + l * DM, ln2_b + l * DM, xbuf, DM);

        attn_out += (long)Bsz * Hh * Lc * Lc;

        // distilling conv + BN + ELU + maxpool
        if (l < 2) {
            im2col3_f16<<<dim3((unsigned)(((long)Mr * 3 * DM + 255) / 256)), 256, 0, stream>>>(
                xbuf, a16, Bsz, Lc, DM);
            packB(conv_w + (long)l * 3 * DM * DM, b16, 3 * DM, DM);
            gemm(a16, b16, conv_b + l * DM, nullptr, rbuf, Mr, DM, 3 * DM, ACT_NONE);
            bn_stats<<<dim3((unsigned)DM), 256, 0, stream>>>(rbuf, mu, var, Mr, DM);
            bn_elu<<<dim3((unsigned)(((long)Mr * DM + 255) / 256)), 256, 0, stream>>>(
                rbuf, mu, var, bn_g + l * DM, bn_b + l * DM, (long)Mr * DM, DM);
            maxpool_kernel<<<dim3((unsigned)(((long)Bsz * (Lc / 2) * DM + 255) / 256)), 256, 0, stream>>>(
                rbuf, xbuf, Bsz, Lc, DM);
            Lc /= 2;
        }
    }

    // final LN + flat projection (16 x 65536 @ 65536 x 48, WMMA) + softmax
    add_layernorm<<<dim3((unsigned)(Bsz * Lc)), 256, 0, stream>>>(
        xbuf, nullptr, lnf_g, lnf_b, xbuf, DM);
    long Kp = (long)Lc * DM;   // 65536
    cvt(xbuf, a16, (long)Bsz * Kp);
    packB(proj_w, b16, (int)Kp, 48);
    gemm(a16, b16, proj_b, nullptr, logits, Bsz, 48, (int)Kp, ACT_NONE);
    softmax_rows<<<dim3((unsigned)Bsz), 64, 0, stream>>>(logits, out, 48);
}
